// DynamicBatchScheduler_13932873908612
// MI455X (gfx1250) — matrix operand order repaired and checked by reading, not compiled, not run on hardware
//
#include <hip/hip_runtime.h>
#include <hip/hip_bf16.h>

// Problem constants (from reference: N=256, S=512, H=1024)
#define N_IN 256
#define SEQ  512
#define HID  1024
#define K2   512     // H/2, hidden width of the pair MLP
#define FDIM 1024    // fused left(512)+right(512) feature dim

typedef __attribute__((ext_vector_type(16))) __bf16        v16bf;
typedef __attribute__((ext_vector_type(8)))  float         v8f;
typedef __attribute__((ext_vector_type(4)))  unsigned int  u32x4;
typedef __attribute__((ext_vector_type(8)))  unsigned int  u32x8;

static __device__ __forceinline__ unsigned short f2bf(float f) {
    // round-to-nearest-even f32 -> bf16
    unsigned int u = __float_as_uint(f);
    u += 0x7FFFu + ((u >> 16) & 1u);
    return (unsigned short)(u >> 16);
}

// ---------------------------------------------------------------------------
// Kernel A: avg over sequence dim + convert to bf16.  This is the only kernel
// that touches the 512 MB input -> pure streaming-bandwidth kernel.
// grid = N_IN blocks, 256 threads; each thread owns 4 hidden channels.
// ---------------------------------------------------------------------------
__global__ void mean_bf16_kernel(const float* __restrict__ in,
                                 unsigned short* __restrict__ avgb) {
    const int n = blockIdx.x;
    const int t = threadIdx.x;                         // 0..255 -> 4 floats each
    const float4* p = (const float4*)(in + (size_t)n * SEQ * HID);
    float4 acc = {0.f, 0.f, 0.f, 0.f};
    #pragma unroll 4
    for (int s = 0; s < SEQ; ++s) {
        float4 v = p[(size_t)s * (HID / 4) + t];       // 512B contiguous per wave
        acc.x += v.x; acc.y += v.y; acc.z += v.z; acc.w += v.w;
    }
    const float inv = 1.0f / (float)SEQ;
    uint2 o;
    o.x = (unsigned)f2bf(acc.x * inv) | ((unsigned)f2bf(acc.y * inv) << 16);
    o.y = (unsigned)f2bf(acc.z * inv) | ((unsigned)f2bf(acc.w * inv) << 16);
    ((uint2*)(avgb + (size_t)n * HID))[t] = o;
}

// ---------------------------------------------------------------------------
// Kernel W: repack W1[512][2048] (f32) into Wb[1024][1024] (bf16):
//   Wb[f][h] = (f < 512) ? W1[f][h] : W1[f-512][1024+h]
// so left and right projections become one fused GEMM.
// grid = FDIM blocks, 256 threads (4 elems each).
// ---------------------------------------------------------------------------
__global__ void w1_bf16_kernel(const float* __restrict__ W1,
                               unsigned short* __restrict__ wb) {
    const int f = blockIdx.x;
    const int t = threadIdx.x;
    const float* src = (f < K2) ? (W1 + (size_t)f * (2 * HID))
                                : (W1 + (size_t)(f - K2) * (2 * HID) + HID);
    float4 v = ((const float4*)src)[t];
    uint2 o;
    o.x = (unsigned)f2bf(v.x) | ((unsigned)f2bf(v.y) << 16);
    o.y = (unsigned)f2bf(v.z) | ((unsigned)f2bf(v.w) << 16);
    ((uint2*)(wb + (size_t)f * HID))[t] = o;
}

// ---------------------------------------------------------------------------
// Kernel B: LR[256][1024] = avg[256][1024] @ Wb[1024][1024]^T via
// v_wmma_f32_16x16x32_bf16.  One 16x16 output tile per wave, K-loop of 32.
// VGPR layouts per CDNA5 ISA 7.12.2:
//   A (16x32 bf16): lane m=lane&15; elems a[0..7] -> k = 8*half+0..7,
//                   a[8..15] -> k = 16 + 8*half + 0..7    (two b128 loads)
//   B (32x16 bf16): lane n=lane&15; elems b[0..7] -> k = 8*half+0..7,
//                   b[8..15] -> k = 16 + 8*half + 0..7    (two b128 loads)
//   C/D (16x16 f32): c[r] -> row m = r + 8*half, col n = lane&15
// grid = 128 blocks x 256 threads (8 waves) = 1024 tiles.
// ---------------------------------------------------------------------------
__global__ void gemm_wmma_kernel(const unsigned short* __restrict__ avgb,
                                 const unsigned short* __restrict__ wb,
                                 float* __restrict__ LR) {
    const int wave = threadIdx.x >> 5;
    const int lane = threadIdx.x & 31;
    const int tile = blockIdx.x * 8 + wave;        // 0..1023
    const int i0   = (tile >> 6) << 4;             // 16 tiles over batch rows
    const int f0   = (tile & 63) << 4;             // 64 tiles over features
    const int hsel = lane >> 4;                    // which half-wave
    const int l16  = lane & 15;

    const unsigned short* arow = avgb + (size_t)(i0 + l16) * HID;
    const unsigned short* brow = wb   + (size_t)(f0 + l16) * HID;

    v8f c = {};
    for (int kk = 0; kk < HID; kk += 32) {
        const int abase = kk + (hsel << 3);        // byte offset 16B-aligned
        u32x8 aw;
        aw.lo = *(const u32x4*)(arow + abase);       // k = abase..abase+7
        aw.hi = *(const u32x4*)(arow + abase + 16);  // k = abase+16..abase+23
        const int bbase = kk + (hsel << 3);        // byte offset 16B-aligned
        u32x8 bw;
        bw.lo = *(const u32x4*)(brow + bbase);       // k = bbase..bbase+7
        bw.hi = *(const u32x4*)(brow + bbase + 16);  // k = bbase+16..bbase+23
        v16bf av = __builtin_bit_cast(v16bf, aw);
        v16bf bv = __builtin_bit_cast(v16bf, bw);
        c = __builtin_amdgcn_wmma_f32_16x16x32_bf16(
                /*neg_a=*/false, av, /*neg_b=*/false, bv,
                /*c_mod=*/(short)0, c, /*reuse_a=*/false, /*reuse_b=*/false);
    }
    #pragma unroll
    for (int r = 0; r < 8; ++r) {
        const int i = i0 + r + (hsel << 3);
        LR[(size_t)i * FDIM + f0 + l16] = c[r];
    }
}

// ---------------------------------------------------------------------------
// Kernel C: out[i][j] = (i==j) ? 0 : sigmoid(sum_k relu(L[i,k]+R[j,k]+b1[k])*W2[k] + b2)
// 16x16 pair tile per block; left/right K-chunks staged in LDS with a
// 260-float padded row stride (260 % 64 == 4 -> conflict-free reads: lanes
// hit 16 distinct banks for L, 2-address broadcast for R).
// grid = 256 blocks x 256 threads.
// ---------------------------------------------------------------------------
__global__ void pair_kernel(const float* __restrict__ LR,
                            const float* __restrict__ b1,
                            const float* __restrict__ W2,
                            const float* __restrict__ b2,
                            float* __restrict__ out) {
    const int KC = 256;                        // K chunk staged per pass
    __shared__ float sL[16 * 260];
    __shared__ float sR[16 * 260];
    __shared__ float sW2[K2];
    __shared__ float sB1[K2];

    const int t  = threadIdx.x;
    const int i0 = (blockIdx.x >> 4) << 4;
    const int j0 = (blockIdx.x & 15) << 4;

    for (int idx = t; idx < K2; idx += 256) { sW2[idx] = W2[idx]; sB1[idx] = b1[idx]; }

    const int ti = t & 15;                     // pair row within tile
    const int tj = t >> 4;                     // pair col within tile
    float acc = 0.0f;

    for (int kc = 0; kc < K2; kc += KC) {
        __syncthreads();
        for (int idx = t; idx < 16 * KC; idx += 256) {
            const int r = idx >> 8;            // 0..15
            const int c = idx & (KC - 1);      // coalesced global reads
            sL[r * 260 + c] = LR[(size_t)(i0 + r) * FDIM + kc + c];
            sR[r * 260 + c] = LR[(size_t)(j0 + r) * FDIM + K2 + kc + c];
        }
        __syncthreads();
        const float* lrow = sL + ti * 260;
        const float* rrow = sR + tj * 260;
        #pragma unroll 8
        for (int k = 0; k < KC; ++k) {
            const float hv = lrow[k] + rrow[k] + sB1[kc + k];
            acc += fmaxf(hv, 0.0f) * sW2[kc + k];
        }
    }

    const float x   = acc + b2[0];
    const float sim = 1.0f / (1.0f + __expf(-x));
    const int i = i0 + ti, j = j0 + tj;
    out[i * N_IN + j] = (i == j) ? 0.0f : sim;
}

// ---------------------------------------------------------------------------
extern "C" void kernel_launch(void* const* d_in, const int* in_sizes, int n_in,
                              void* d_out, int out_size, void* d_ws, size_t ws_size,
                              hipStream_t stream) {
    const float* feats = (const float*)d_in[0];   // [256,512,1024]
    const float* W1    = (const float*)d_in[1];   // [512,2048]
    const float* b1    = (const float*)d_in[2];   // [512]
    const float* W2    = (const float*)d_in[3];   // [1,512] flat
    const float* b2    = (const float*)d_in[4];   // [1]
    float*       out   = (float*)d_out;           // [256,256]

    char* ws = (char*)d_ws;
    unsigned short* avgb = (unsigned short*)ws;                            // 512 KB
    unsigned short* wb   = (unsigned short*)(ws + 512 * 1024);             // 2 MB
    float*          LR   = (float*)(ws + 512 * 1024 + 2 * 1024 * 1024);    // 1 MB

    mean_bf16_kernel<<<N_IN, 256, 0, stream>>>(feats, avgb);
    w1_bf16_kernel  <<<FDIM, 256, 0, stream>>>(W1, wb);
    gemm_wmma_kernel<<<128, 256, 0, stream>>>(avgb, wb, LR);
    pair_kernel     <<<256, 256, 0, stream>>>(LR, b1, W2, b2, out);
}
